// Attention_60275571032317
// MI455X (gfx1250) — matrix operand order repaired and checked by reading, not compiled, not run on hardware
//
#include <hip/hip_runtime.h>

// ---------------------------------------------------------------------------
// Derivation note: reference computes softmax_j(cq_i + ck_j + bias_A). Both
// cq_i and bias_A are constant in the softmax axis j, so they cancel exactly
// (softmax shift invariance). attn = softmax_j(ck_j), independent of i, and
// out[..., i] = sum_j attn_j * v_j is broadcast over i. Only conv(key) is
// needed; query and conv_b are mathematically unused.
//
// conv(key) is an implicit GEMM on v_wmma_f32_16x16x32_bf16 with split-bf16
// (hi+lo) operands: hi*whi + hi*wlo + lo*whi accumulated in f32 (~fp32
// accuracy at bf16 matrix throughput). Weight fragments are DMA'd into LDS
// once per workgroup via the Tensor Data Mover (tensor_load_to_lds +
// s_wait_tensorcnt), then served per-lane with ds_load_b128.
// ---------------------------------------------------------------------------

typedef __attribute__((ext_vector_type(16))) __bf16   v16bf;
typedef __attribute__((ext_vector_type(8)))  float    v8f;
typedef __attribute__((ext_vector_type(4)))  int      i4;    // 16B chunk
typedef __attribute__((ext_vector_type(4)))  float    f4;
typedef __attribute__((ext_vector_type(4)))  unsigned u32x4;
typedef __attribute__((ext_vector_type(8)))  int      i32x8;
typedef __attribute__((ext_vector_type(4)))  int      i32x4;

union Frag { v16bf v; i4 q[2]; };

constexpr int BB = 8, NN = 128, HH = 32, WW = 32, LL = 16, HEADS = 8;
constexpr int POS = HH * WW;          // 1024
constexpr int NCHUNK = 100;           // 25 taps * 4 ci-chunks of 32

// workspace layout (bytes); wfrag hi and lo are contiguous so one TDM
// descriptor can fetch both (100*512*2 = 102400 bytes each).
constexpr size_t OFF_WH = 0;
constexpr size_t OFF_WL = 102400;
constexpr size_t WFRAG_BYTES = 204800;                        // hi + lo
constexpr size_t OFF_KH = 204800;                             // 128*1024*128*2 = 32 MiB
constexpr size_t OFF_KL = OFF_KH + 33554432ull;
constexpr size_t OFF_CF = OFF_KL + 33554432ull;               // ckf: 8*8*1024*16*4 = 4 MiB

// ---------------------------------------------------------------------------
// Kernel 1: build WMMA B-fragments of conv_w (hi/lo bf16).
// B layout (32x16 bf16): lane L -> column n = L%16; element j -> K =
// (L<16 ? 0 : 8) + j + (j<8 ? 0 : 8). Chunk c: tap = c/4, ci base = (c%4)*32.
// B[k][n] = w[n][ci_base+k][tap]  (n >= 8 padded with zeros)
// ---------------------------------------------------------------------------
__global__ void stage_w_kernel(const float* __restrict__ w,
                               __bf16* __restrict__ wh, __bf16* __restrict__ wl) {
  int chunk = blockIdx.x;            // 0..99
  int lane  = threadIdx.x;           // 0..31
  int tap   = chunk >> 2;
  int cib   = (chunk & 3) * 32;
  int n     = lane & 15;
  int kbase = (lane < 16) ? 0 : 8;

  alignas(32) __bf16 hi[16];
  alignas(32) __bf16 lo[16];
#pragma unroll
  for (int j = 0; j < 16; ++j) {
    float v = 0.0f;
    if (n < HEADS) v = w[(n * NN + (cib + kbase + j + (j < 8 ? 0 : 8))) * 25 + tap];
    __bf16 h = (__bf16)v;
    hi[j] = h;
    lo[j] = (__bf16)(v - (float)h);
  }
  __bf16* dh = wh + ((size_t)chunk * 512 + lane * 16);
  __bf16* dl = wl + ((size_t)chunk * 512 + lane * 16);
  ((i4*)dh)[0] = *(i4*)&hi[0];  ((i4*)dh)[1] = *(i4*)&hi[8];
  ((i4*)dl)[0] = *(i4*)&lo[0];  ((i4*)dl)[1] = *(i4*)&lo[8];
}

// ---------------------------------------------------------------------------
// Kernel 2: stage key fp32 [b][ci][y][x][l] -> bf16 hi/lo NHWC per image:
// kb[img = b*16 + l][y][x][ci]. LDS transpose so both reads (one 64B line per
// thread) and writes (32B runs) are coalesced.
// ---------------------------------------------------------------------------
__global__ void stage_k_kernel(const float* __restrict__ key,
                               __bf16* __restrict__ kh, __bf16* __restrict__ kl) {
  __shared__ alignas(16) __bf16 shh[16][128];
  __shared__ alignas(16) __bf16 shl[16][128];

  int blk = blockIdx.x;              // b*1024 + y*32 + x
  int b = blk >> 10, y = (blk >> 5) & 31, x = blk & 31;
  int ci = threadIdx.x;              // 0..127

  const float* p = key + ((((size_t)b * NN + ci) * HH + y) * WW + x) * LL;
  f4 r0 = ((const f4*)p)[0];
  f4 r1 = ((const f4*)p)[1];
  f4 r2 = ((const f4*)p)[2];
  f4 r3 = ((const f4*)p)[3];

#pragma unroll
  for (int l = 0; l < 16; ++l) {
    float v = (l < 4 ? r0[l & 3] : l < 8 ? r1[l & 3] : l < 12 ? r2[l & 3] : r3[l & 3]);
    __bf16 h = (__bf16)v;
    shh[l][ci] = h;
    shl[l][ci] = (__bf16)(v - (float)h);
  }
  __syncthreads();

  int l = threadIdx.x >> 3;          // 0..15
  int cg = threadIdx.x & 7;          // 0..7 -> 16-channel group
  size_t img = (size_t)b * 16 + l;
  size_t o = (img * POS + (size_t)y * WW + x) * NN + cg * 16;
  i4* sh0 = (i4*)&shh[l][cg * 16];
  i4* sl0 = (i4*)&shl[l][cg * 16];
  ((i4*)(kh + o))[0] = sh0[0];  ((i4*)(kh + o))[1] = sh0[1];
  ((i4*)(kl + o))[0] = sl0[0];  ((i4*)(kl + o))[1] = sl0[1];
}

// ---------------------------------------------------------------------------
// Kernel 3: conv as implicit GEMM on WMMA bf16 (split hi/lo, 3 WMMAs/chunk).
// Block = 256 threads = 8 waves; grid = 128 images * 2 halves; 204800 B of
// dynamic LDS holds all 100 K-chunk B-fragments (hi then lo), filled once by
// the Tensor Data Mover (wave 0 issues the 1-D descriptor; TDM ignores EXEC),
// completed with s_wait_tensorcnt then a workgroup barrier.
//
// A layout (16x32 bf16): lane L -> row m = L%16; elems 0..7 = K ko..ko+7,
// elems 8..15 = K ko+16..ko+23, ko = (L<16 ? 0 : 8). dy boundary skip is
// wave-uniform (EXEC stays full for WMMA); dx boundary -> per-lane zero frag.
// Output ckf[b][head][pos][l] fp32 from the C/D layout: lane L -> col n=L%16,
// elem v -> row m = v + (L<16 ? 0 : 8).
// ---------------------------------------------------------------------------
__global__ __launch_bounds__(256) void conv_wmma_kernel(
    const __bf16* __restrict__ kh, const __bf16* __restrict__ kl,
    const __bf16* __restrict__ wfrag,   // hi at +0, lo at +102400 bytes
    float* __restrict__ ckf) {
  extern __shared__ __align__(16) char smem[];    // 204800 B (wfrag hi+lo)
  __bf16* swf = (__bf16*)smem;

  int img  = blockIdx.x >> 1;        // 0..127  (= b*16 + l)
  int half = blockIdx.x & 1;
  int wave = threadIdx.x >> 5;
  int lane = threadIdx.x & 31;
  int m    = lane & 15;
  int ko   = (lane < 16) ? 0 : 8;
  int n    = lane & 15;
  int bimg = img >> 4;
  int lstp = img & 15;

#if __has_builtin(__builtin_amdgcn_tensor_load_to_lds) && \
    __has_builtin(__builtin_amdgcn_s_wait_tensorcnt)
  if (wave == 0) {
    // Tensor DMA descriptor (D#), 1-D: 51200 x 4B elements, global -> LDS.
    unsigned long long ga = (unsigned long long)(uintptr_t)wfrag;
    unsigned lds_off = (unsigned)(uintptr_t)smem;   // low 32 bits = LDS offset
    // Group 0: count=1 | lds_addr | global_addr[31:0] |
    //          global_addr[56:32] + type=2 ("image") in bits 127:126
    u32x4 g0 = { 1u, lds_off, (unsigned)ga,
                 (unsigned)((ga >> 32) & 0x01FFFFFFu) | 0x80000000u };
    // Group 1: data_size=2 (4B) @bits17:16; tensor_dim0=51200 @bits79:48;
    //          tensor_dim1=1 @bits111:80; tile_dim0=51200 @bits127:112;
    //          tile_dim1=1 @bits143:128; tensor_dim0_stride=51200 @bits207:160
    i32x8 g1 = { 0x00020000, (int)0xC8000000u, 0x00010000, (int)0xC8000000u,
                 1, 51200, 0, 0 };
    i32x4 g2 = { 0, 0, 0, 0 };
    i32x4 g3 = { 0, 0, 0, 0 };
    i32x8 g4 = { 0, 0, 0, 0, 0, 0, 0, 0 };   // 6-arg toolchain form
    __builtin_amdgcn_tensor_load_to_lds(g0, g1, g2, g3, g4, 0);
    __builtin_amdgcn_s_wait_tensorcnt((short)0);
  }
#else
  // Fallback: cooperative vector copy global -> LDS.
  for (int ofs = threadIdx.x * 16; ofs < (int)WFRAG_BYTES; ofs += 256 * 16) {
    *(i4*)(smem + ofs) = *(const i4*)((const char*)wfrag + ofs);
  }
#endif
  __syncthreads();

  const __bf16* swh = swf;                         // hi fragments
  const __bf16* swl = swf + (OFF_WL / 2);          // lo fragments (bf16 idx)
  const __bf16* khi = kh + (size_t)img * POS * NN;
  const __bf16* klo = kl + (size_t)img * POS * NN;

  for (int s = 0; s < 4; ++s) {
    int T  = half * 32 + wave * 4 + s;   // 0..63, wave-uniform
    int y  = T >> 1;
    int x0 = (T & 1) * 16;
    v8f acc = {0.f, 0.f, 0.f, 0.f, 0.f, 0.f, 0.f, 0.f};

    for (int dy = 0; dy < 5; ++dy) {
      int yy = y + dy - 2;
      if (yy < 0 || yy >= HH) continue;          // uniform across the wave
      for (int dx = 0; dx < 5; ++dx) {
        int xx = x0 + m + dx - 2;                 // per-lane
        bool vx = (xx >= 0) && (xx < WW);
        int rowoff = (yy * WW + (vx ? xx : 0)) * NN;
        int tap = dy * 5 + dx;
#pragma unroll
        for (int cc = 0; cc < 4; ++cc) {
          int chunk = tap * 4 + cc;
          int cib = cc * 32;
          i4 z = {0, 0, 0, 0};
          Frag Ah, Al, Bh, Bl;
          Ah.q[0] = z; Ah.q[1] = z; Al.q[0] = z; Al.q[1] = z;
          if (vx) {
            const __bf16* ap = khi + rowoff + cib + ko;
            const __bf16* bp = klo + rowoff + cib + ko;
            Ah.q[0] = *(const i4*)ap;  Ah.q[1] = *(const i4*)(ap + 16);
            Al.q[0] = *(const i4*)bp;  Al.q[1] = *(const i4*)(bp + 16);
          }
          const __bf16* wph = swh + chunk * 512 + lane * 16;   // LDS, 32B/lane
          const __bf16* wpl = swl + chunk * 512 + lane * 16;
          Bh.q[0] = ((const i4*)wph)[0];  Bh.q[1] = ((const i4*)wph)[1];
          Bl.q[0] = ((const i4*)wpl)[0];  Bl.q[1] = ((const i4*)wpl)[1];

          acc = __builtin_amdgcn_wmma_f32_16x16x32_bf16(
              false, Ah.v, false, Bh.v, (short)0, acc, false, false);
          acc = __builtin_amdgcn_wmma_f32_16x16x32_bf16(
              false, Ah.v, false, Bl.v, (short)0, acc, false, false);
          acc = __builtin_amdgcn_wmma_f32_16x16x32_bf16(
              false, Al.v, false, Bh.v, (short)0, acc, false, false);
        }
      }
    }

    if (n < HEADS) {
      int mb = (lane < 16) ? 0 : 8;
      int pos0 = T * 16;
      size_t base = (((size_t)bimg * HEADS + n) * POS) * LL + lstp;
#pragma unroll
      for (int v = 0; v < 8; ++v) {
        ckf[base + (size_t)(pos0 + mb + v) * LL] = acc[v];
      }
    }
  }
}

// ---------------------------------------------------------------------------
// Kernel 4: attn = softmax_l(ckf), out[b][A*16+cc][pos][i] = dot(attn, v)
// broadcast over all i. One thread per (b, head, pos); 64B/lane coalesced.
// ---------------------------------------------------------------------------
__global__ __launch_bounds__(256) void softmax_av_kernel(
    const float* __restrict__ ckf, const float* __restrict__ value,
    float* __restrict__ out) {
  int t   = blockIdx.x * 256 + threadIdx.x;   // 0..65535
  int pos = t & 1023;
  int A   = (t >> 10) & 7;
  int b   = t >> 13;

  const float* cp = ckf + (((size_t)b * 8 + A) * POS + pos) * 16;
  f4 c0 = ((const f4*)cp)[0];
  f4 c1 = ((const f4*)cp)[1];
  f4 c2 = ((const f4*)cp)[2];
  f4 c3 = ((const f4*)cp)[3];

  float logit[16];
#pragma unroll
  for (int j = 0; j < 16; ++j)
    logit[j] = (j < 4 ? c0[j & 3] : j < 8 ? c1[j & 3] : j < 12 ? c2[j & 3] : c3[j & 3]);

  float mx = logit[0];
#pragma unroll
  for (int j = 1; j < 16; ++j) mx = fmaxf(mx, logit[j]);
  float sum = 0.0f;
  float attn[16];
#pragma unroll
  for (int j = 0; j < 16; ++j) { attn[j] = __expf(logit[j] - mx); sum += attn[j]; }
  float inv = 1.0f / sum;
#pragma unroll
  for (int j = 0; j < 16; ++j) attn[j] *= inv;

#pragma unroll 4
  for (int cc = 0; cc < 16; ++cc) {
    size_t off = (((size_t)b * NN + A * 16 + cc) * POS + pos) * 16;
    const f4* vp = (const f4*)(value + off);
    f4 v0 = vp[0], v1 = vp[1], v2 = vp[2], v3 = vp[3];
    float o = 0.0f;
#pragma unroll
    for (int e = 0; e < 4; ++e) {
      o = fmaf(attn[e],      v0[e], o);
      o = fmaf(attn[4 + e],  v1[e], o);
      o = fmaf(attn[8 + e],  v2[e], o);
      o = fmaf(attn[12 + e], v3[e], o);
    }
    f4 ov = {o, o, o, o};
    f4* op = (f4*)(out + off);
    op[0] = ov; op[1] = ov; op[2] = ov; op[3] = ov;   // broadcast over i
  }
}

// ---------------------------------------------------------------------------
extern "C" void kernel_launch(void* const* d_in, const int* in_sizes, int n_in,
                              void* d_out, int out_size, void* d_ws, size_t ws_size,
                              hipStream_t stream) {
  (void)in_sizes; (void)n_in; (void)out_size; (void)ws_size;
  // inputs: 0=query (unused: cancels in softmax), 1=key, 2=value,
  //         3=conv_w, 4=conv_b (unused: cancels in softmax)
  const float* key    = (const float*)d_in[1];
  const float* value  = (const float*)d_in[2];
  const float* conv_w = (const float*)d_in[3];
  float* out = (float*)d_out;

  char* ws = (char*)d_ws;                       // needs ~71.5 MB
  __bf16* wh  = (__bf16*)(ws + OFF_WH);
  __bf16* wl  = (__bf16*)(ws + OFF_WL);
  __bf16* kh  = (__bf16*)(ws + OFF_KH);
  __bf16* kl  = (__bf16*)(ws + OFF_KL);
  float*  ckf = (float*)(ws + OFF_CF);

  stage_w_kernel<<<NCHUNK, 32, 0, stream>>>(conv_w, wh, wl);
  stage_k_kernel<<<BB * POS, 128, 0, stream>>>(key, kh, kl);
  conv_wmma_kernel<<<BB * LL * 2, 256, (int)WFRAG_BYTES, stream>>>(kh, kl, wh, ckf);
  softmax_av_kernel<<<(BB * HEADS * POS) / 256, 256, 0, stream>>>(ckf, value, out);
}
